// LocalGlobalDiscriminator_64991445123092
// MI455X (gfx1250) — compile-verified, matrix operand order checked by reading, not run on hardware
//
#include <hip/hip_runtime.h>

// ---------------------------------------------------------------------------
// CDNA5 (gfx1250) LocalGlobalDiscriminator forward pass.
// wave32; v_wmma_f32_16x16x32_f16 for all GEMM-shaped work.
// All intermediate tensors are POINT-MAJOR (P, C) so that:
//   - WMMA B fragments load as two global_load_b128 per lane (K-pairs contig)
//   - WMMA D tiles store as one global_store_b128 per lane (8 consecutive M)
// Intermediates are f16 (BN renormalizes; halves HBM traffic).
// ---------------------------------------------------------------------------

typedef _Float16 half_t;
typedef __attribute__((ext_vector_type(16))) _Float16 v16h;
typedef __attribute__((ext_vector_type(8)))  _Float16 v8h;
typedef __attribute__((ext_vector_type(8)))  float    v8f;

#define BN_EPS 1e-5f
static constexpr int Bc = 2;     // batch
static constexpr int Np = 8192;  // points

// K index base for 16-bit A/B WMMA operand layout (16x32 / 32x16):
// VGPR v (0..7), lane-half h: K = (v&4 ? 16:0) + h*8 + (v&3)*2
__device__ __forceinline__ int wmma_kb(int v, int half) {
  return ((v & 4) ? 16 : 0) + half * 8 + ((v & 3) << 1);
}

// Pack 32 K-values worth of one lane's fragment (8 floats at p[0..7] and 8 at
// p[16..23], 16B-aligned) into the v16h operand register block.
__device__ __forceinline__ v16h pack_f32_frag(const float* p) {
  float4 q0 = *(const float4*)(p);
  float4 q1 = *(const float4*)(p + 4);
  float4 q2 = *(const float4*)(p + 16);
  float4 q3 = *(const float4*)(p + 20);
  v16h f;
  f[0] = (half_t)q0.x; f[1] = (half_t)q0.y; f[2]  = (half_t)q0.z; f[3]  = (half_t)q0.w;
  f[4] = (half_t)q1.x; f[5] = (half_t)q1.y; f[6]  = (half_t)q1.z; f[7]  = (half_t)q1.w;
  f[8] = (half_t)q2.x; f[9] = (half_t)q2.y; f[10] = (half_t)q2.z; f[11] = (half_t)q2.w;
  f[12] = (half_t)q3.x; f[13] = (half_t)q3.y; f[14] = (half_t)q3.z; f[15] = (half_t)q3.w;
  return f;
}

__device__ __forceinline__ void atomicMaxF(float* addr, float val) {
  unsigned* ua = (unsigned*)addr;
  unsigned cur = *ua;
  while (__uint_as_float(cur) < val) {
    unsigned prev = atomicCAS(ua, cur, __float_as_uint(val));
    if (prev == cur) break;
    cur = prev;
  }
}

// ---------------------------------------------------------------------------
// Generic WMMA GEMM: Y(M,P) = W(M,K) * F(K,P), f16 compute / f32 accumulate.
// BMODE = how column p of F is sourced; OMODE = output path.
// ---------------------------------------------------------------------------
struct GemmArgs {
  const float*  W;     // M x K (f32 weights)
  const float*  bias;  // len M or null (OMODE 1 only)
  const float*  srcA;  // primary f32 source (point-major)
  const float*  srcB;  // secondary f32 source (point-major feats)
  const float*  srcC;  // centroids xyz (point-major)
  const half_t* srcH;  // f16 source, layout (P, K), K % 32 == 0 (BMODE 0)
  const int*    idx;   // gather indices
  half_t* outH;        // OMODE 0: (P, M)
  float*  outF;        // OMODE 1: (B, Nout, M), init -3e38, atomic max
  int M, K, P, C, N, S, kE, Nout, g;
};

// Per-lane B-column context, computed once before the K loop.
struct BCtx {
  const float*  fp;   // row pointer 1
  const float*  fp2;  // row pointer 2
  const half_t* hp;   // f16 row pointer
  float x0, x1, x2;   // xyz values (SA modes)
};

// BMODE 0: dense f16 (P,K);            BMODE 1: edge feats [nbr-ctr ; ctr]
// BMODE 2: SA group [xyz[pt]-new ; feat[pt]]; BMODE 3: SA-all [xyz2 ; f2]
// BMODE 4: dense f32 point-major (B*N, K)
template <int BMODE>
__device__ __forceinline__ BCtx prepB(const GemmArgs& A, int p) {
  BCtx c{};
  if constexpr (BMODE == 0) {
    c.hp = A.srcH + (size_t)p * A.K;
  } else if constexpr (BMODE == 1) {
    int NK = A.N * A.kE;
    int b = p / NK; int r = p - b * NK;
    int n = r / A.kE; int j = r - n * A.kE;
    const float* base = A.srcA + (size_t)b * A.N * A.C;
    int pt = A.idx[((size_t)b * A.N + n) * A.kE + j];
    c.fp  = base + (size_t)pt * A.C;  // neighbor row
    c.fp2 = base + (size_t)n * A.C;   // center row
  } else if constexpr (BMODE == 2) {
    int SK = A.S * A.kE;
    int b = p / SK; int r = p - b * SK;
    int s = r / A.kE; int j = r - s * A.kE;
    int pt = A.idx[((size_t)b * A.S + s) * A.kE + j];
    const float* xp = A.srcA + ((size_t)b * A.N + pt) * 3;
    const float* cp = A.srcC + ((size_t)b * A.S + s) * 3;
    c.x0 = xp[0] - cp[0]; c.x1 = xp[1] - cp[1]; c.x2 = xp[2] - cp[2];
    c.fp = A.srcB + ((size_t)b * A.N + pt) * A.C;
  } else if constexpr (BMODE == 3) {
    int b = p / A.S; int s = p - b * A.S;
    const float* xp = A.srcA + ((size_t)b * A.S + s) * 3;
    c.x0 = xp[0]; c.x1 = xp[1]; c.x2 = xp[2];
    c.fp = A.srcB + ((size_t)b * A.S + s) * A.C;
  } else {
    int b = p / A.N; int n = p - b * A.N;
    c.fp = A.srcA + ((size_t)b * A.N + n) * A.K;
  }
  return c;
}

template <int BMODE>
__device__ __forceinline__ float loadB(const GemmArgs& A, const BCtx& c, int kk) {
  if constexpr (BMODE == 1) {
    return kk < A.C ? c.fp[kk] - c.fp2[kk] : c.fp2[kk - A.C];
  } else if constexpr (BMODE == 2) {
    if (kk >= 3) return c.fp[kk - 3];
    return kk == 0 ? c.x0 : (kk == 1 ? c.x1 : c.x2);
  } else if constexpr (BMODE == 3) {
    if (kk >= 3) return c.fp[kk - 3];
    return kk == 0 ? c.x0 : (kk == 1 ? c.x1 : c.x2);
  } else {  // BMODE 4
    return c.fp[kk];
  }
}

template <int BMODE, int OMODE>
__global__ void gemm_wmma_kernel(GemmArgs A) {
  const int wave = threadIdx.x >> 5, lane = threadIdx.x & 31;
  const int half = lane >> 4, l16 = lane & 15;
  const int m0 = blockIdx.y * 16;
  const int p0 = blockIdx.x * 64 + wave * 16;
  const int row = m0 + l16;   // A row (M)
  const int p   = p0 + l16;   // B column (P)

  const float* wr = A.W + (size_t)row * A.K;
  const BCtx ctx = prepB<BMODE>(A, p);

  v8f acc = {};
  for (int kc = 0; kc < A.K; kc += 32) {
    __builtin_prefetch(&wr[kc + 32], 0, 0);
    v16h av, bv;
#pragma unroll
    for (int v = 0; v < 8; v++) {
      int kb = kc + wmma_kb(v, half);
      av[2 * v]     = (half_t)((kb     < A.K) ? wr[kb]     : 0.f);
      av[2 * v + 1] = (half_t)((kb + 1 < A.K) ? wr[kb + 1] : 0.f);
    }
    if constexpr (BMODE == 0) {
      // K % 32 == 0 here: two 16B loads cover the lane's 16 f16 values.
      const half_t* hp = ctx.hp + kc + half * 8;
      v8h lo = *(const v8h*)(hp);
      v8h hi = *(const v8h*)(hp + 16);
#pragma unroll
      for (int t = 0; t < 8; t++) { bv[t] = lo[t]; bv[8 + t] = hi[t]; }
    } else {
#pragma unroll
      for (int v = 0; v < 8; v++) {
        int kb = kc + wmma_kb(v, half);
        bv[2 * v]     = (half_t)((kb     < A.K) ? loadB<BMODE>(A, ctx, kb)     : 0.f);
        bv[2 * v + 1] = (half_t)((kb + 1 < A.K) ? loadB<BMODE>(A, ctx, kb + 1) : 0.f);
      }
    }
    acc = __builtin_amdgcn_wmma_f32_16x16x32_f16(false, av, false, bv,
                                                 (short)0, acc, false, false);
  }

  if constexpr (OMODE == 0) {
    // lane holds rows m0+half*8 .. +7 of column p: one contiguous b128 store
    v8h hv;
#pragma unroll
    for (int r = 0; r < 8; r++) hv[r] = (half_t)acc[r];
    *(v8h*)(A.outH + (size_t)p * A.M + m0 + half * 8) = hv;
  } else {
    const int rw = A.g < 16 ? A.g : 16;  // in-tile group width (8 or 16)
#pragma unroll
    for (int r = 0; r < 8; r++) {
      int m = m0 + r + half * 8;
      float val = acc[r] + (A.bias ? A.bias[m] : 0.f);
      for (int off = 1; off < rw; off <<= 1)
        val = fmaxf(val, __shfl_xor(val, off, 32));
      if ((l16 & (rw - 1)) == 0) {
        int b = p / (A.Nout * A.g);
        int n = (p / A.g) % A.Nout;
        atomicMaxF(&A.outF[((size_t)b * A.Nout + n) * A.M + m], val);
      }
    }
  }
}

// ---------------------------------------------------------------------------
// Fused feature-KNN: WMMA Gram tiles + in-LDS top-k; X point-major (B,N,C),
// C in {64,128}. A-fragments hoisted across the candidate loop.
// grid:(N/16, B), block 128.
// ---------------------------------------------------------------------------
__global__ void knn_wmma_kernel(const float* X, const float* norms, int C,
                                int N, int kOut, int* idxOut) {
  const int b = blockIdx.y, m0 = blockIdx.x * 16;
  const int wave = threadIdx.x >> 5, lane = threadIdx.x & 31;
  const int half = lane >> 4, l16 = lane & 15;
  const float* Xb = X + (size_t)b * N * C;
  const float* nb = norms + (size_t)b * N;
  __shared__ float sdist[16][64];

  const int C32 = C >> 5;
  const float* aq = Xb + (size_t)(m0 + l16) * C + half * 8;
  v16h afrag[4];
#pragma unroll
  for (int kci = 0; kci < 4; kci++)
    if (kci < C32) afrag[kci] = pack_f32_frag(aq + kci * 32);

  float nqr[8];
#pragma unroll
  for (int r = 0; r < 8; r++) nqr[r] = nb[m0 + r + half * 8];

  float best[16]; int bidx[16];
  for (int i = 0; i < kOut; i++) { best[i] = -3e38f; bidx[i] = 0; }

  for (int c0 = 0; c0 < N; c0 += 64) {
    const int col = c0 + wave * 16 + l16;
    const float* bq = Xb + (size_t)col * C + half * 8;
    v8f acc = {};
#pragma unroll
    for (int kci = 0; kci < 4; kci++) {
      if (kci < C32) {
        v16h bfrag = pack_f32_frag(bq + kci * 32);
        acc = __builtin_amdgcn_wmma_f32_16x16x32_f16(false, afrag[kci], false,
                                                     bfrag, (short)0, acc,
                                                     false, false);
      }
    }
    float ncol = nb[col];
#pragma unroll
    for (int r = 0; r < 8; r++)
      sdist[r + half * 8][wave * 16 + l16] = 2.f * acc[r] - nqr[r] - ncol;
    __syncthreads();
    if (threadIdx.x < 16) {
      int rowq = threadIdx.x;
      for (int j = 0; j < 64; j++) {
        float s = sdist[rowq][j];
        int wi = 0; float wv = best[0];
        for (int i = 1; i < kOut; i++)
          if (best[i] < wv) { wv = best[i]; wi = i; }
        if (s > wv) { best[wi] = s; bidx[wi] = c0 + j; }
      }
    }
    __syncthreads();
  }
  if (threadIdx.x < 16) {
    int rowq = threadIdx.x;
    for (int i = 0; i < kOut; i++)
      idxOut[((size_t)b * N + m0 + rowq) * kOut + i] = bidx[i];
  }
}

// ---------------------------------------------------------------------------
// 3-dim brute-force KNN (point-major xyz): Q (B,Sq,3), X (B,Nc,3)
// ---------------------------------------------------------------------------
__global__ void knn3_kernel(const float* Q, int Sq, const float* X, int Nc,
                            int k, int* out) {
  int t = blockIdx.x * blockDim.x + threadIdx.x;
  if (t >= Bc * Sq) return;
  int b = t / Sq, s = t - b * Sq;
  const float* q = Q + ((size_t)b * Sq + s) * 3;
  float qx = q[0], qy = q[1], qz = q[2];
  const float* xb = X + (size_t)b * Nc * 3;
  float bd[32]; int bi[32];
  for (int i = 0; i < k; i++) { bd[i] = 3e38f; bi[i] = 0; }
  for (int n = 0; n < Nc; n++) {
    const float* xp = xb + (size_t)n * 3;
    float dx = xp[0] - qx, dy = xp[1] - qy, dz = xp[2] - qz;
    float d = dx * dx + dy * dy + dz * dz;
    int wi = 0; float wv = bd[0];
    for (int i = 1; i < k; i++)
      if (bd[i] > wv) { wv = bd[i]; wi = i; }
    if (d < wv) { bd[wi] = d; bi[wi] = n; }
  }
  for (int i = 0; i < k; i++) out[((size_t)b * Sq + s) * k + i] = bi[i];
}

// ---------------------------------------------------------------------------
// Farthest point sampling on point-major xyz; one block per batch element.
// ---------------------------------------------------------------------------
__global__ void fps_kernel(const float* X, int N, int S, int* fidx,
                           float* mind) {
  int b = blockIdx.x, tid = threadIdx.x;
  const float* xb = X + (size_t)b * N * 3;
  __shared__ float scent[3];
  __shared__ float smax[256];
  __shared__ int   simax[256];
  __shared__ int   scur;
  for (int n = tid; n < N; n += 256) mind[(size_t)b * N + n] = 1e10f;
  if (tid == 0) scur = 0;
  __syncthreads();
  for (int i = 0; i < S; i++) {
    if (tid == 0) {
      fidx[b * S + i] = scur;
      scent[0] = xb[(size_t)scur * 3 + 0];
      scent[1] = xb[(size_t)scur * 3 + 1];
      scent[2] = xb[(size_t)scur * 3 + 2];
    }
    __syncthreads();
    float lm = -1.f; int li = 0;
    for (int n = tid; n < N; n += 256) {
      const float* xp = xb + (size_t)n * 3;
      float dx = xp[0] - scent[0], dy = xp[1] - scent[1], dz = xp[2] - scent[2];
      float d = dx * dx + dy * dy + dz * dz;
      float md = fminf(mind[(size_t)b * N + n], d);
      mind[(size_t)b * N + n] = md;
      if (md > lm) { lm = md; li = n; }
    }
    smax[tid] = lm; simax[tid] = li;
    __syncthreads();
    for (int o = 128; o > 0; o >>= 1) {
      if (tid < o && smax[tid + o] > smax[tid]) {
        smax[tid] = smax[tid + o]; simax[tid] = simax[tid + o];
      }
      __syncthreads();
    }
    if (tid == 0) scur = simax[0];
    __syncthreads();
  }
}

// ---------------------------------------------------------------------------
// BN over point-major f16 (P,M): coalesced grid-stride + LDS ds_add_f32
// accumulators, global atomic add into acc[2*M] (pre-zeroed). M power of 2.
// ---------------------------------------------------------------------------
__global__ void bn_stats_kernel(const half_t* H, int M, size_t total,
                                float* acc) {
  __shared__ float ssum[256], ssq[256];
  for (int i = threadIdx.x; i < M; i += 256) { ssum[i] = 0.f; ssq[i] = 0.f; }
  __syncthreads();
  size_t stride = (size_t)gridDim.x * blockDim.x;
  for (size_t i = (size_t)blockIdx.x * blockDim.x + threadIdx.x; i < total;
       i += stride) {
    float v = (float)H[i];
    int m = (int)(i & (size_t)(M - 1));
    atomicAdd(&ssum[m], v);
    atomicAdd(&ssq[m], v * v);
  }
  __syncthreads();
  for (int m = threadIdx.x; m < M; m += 256) {
    atomicAdd(&acc[m], ssum[m]);
    atomicAdd(&acc[M + m], ssq[m]);
  }
}

__global__ void bn_apply_kernel(half_t* H, int M, int P, const float* acc,
                                int act) {
  size_t total = (size_t)M * P;
  size_t stride = (size_t)gridDim.x * blockDim.x;
  float invP = 1.f / (float)P;
  for (size_t i = (size_t)blockIdx.x * blockDim.x + threadIdx.x; i < total;
       i += stride) {
    int m = (int)(i & (size_t)(M - 1));
    float mean = acc[m] * invP;
    float var = acc[M + m] * invP - mean * mean;
    float v = ((float)H[i] - mean) * rsqrtf(var + BN_EPS);
    if (act == 1) v = fmaxf(v, 0.f);
    else if (act == 2) v = v > 0.f ? v : 0.2f * v;
    H[i] = (half_t)v;
  }
}

// ---------------------------------------------------------------------------
// Small utility kernels
// ---------------------------------------------------------------------------
__global__ void fill_kernel(float* p, size_t n, float v) {
  size_t stride = (size_t)gridDim.x * blockDim.x;
  for (size_t i = (size_t)blockIdx.x * blockDim.x + threadIdx.x; i < n;
       i += stride)
    p[i] = v;
}

__global__ void norms_kernel(const float* X, int C, int N, float* out) {
  int t = blockIdx.x * blockDim.x + threadIdx.x;
  if (t >= Bc * N) return;
  const float* xp = X + (size_t)t * C;
  float s = 0.f;
  for (int c = 0; c < C; c++) { float v = xp[c]; s += v * v; }
  out[t] = s;
}

__global__ void gatherxyz_kernel(const float* X, int Nc, const int* fidx,
                                 int S, float* out) {
  int t = blockIdx.x * blockDim.x + threadIdx.x;
  if (t >= Bc * S * 3) return;
  int b = t / (S * 3); int r = t - b * S * 3;
  int s = r / 3; int c = r - s * 3;
  out[t] = X[((size_t)b * Nc + fidx[b * S + s]) * 3 + c];
}

// max over N per (b, m); H point-major (B*N, M)
__global__ void rowmax_kernel(const half_t* H, int M, int Nper, float* out) {
  __shared__ float sm[256];
  int m = blockIdx.x, b = blockIdx.y;
  float lm = -3e38f;
  for (int n = threadIdx.x; n < Nper; n += 256)
    lm = fmaxf(lm, (float)H[((size_t)b * Nper + n) * M + m]);
  sm[threadIdx.x] = lm;
  __syncthreads();
  for (int o = 128; o > 0; o >>= 1) {
    if (threadIdx.x < o)
      sm[threadIdx.x] = fmaxf(sm[threadIdx.x], sm[threadIdx.x + o]);
    __syncthreads();
  }
  if (threadIdx.x == 0) out[b * M + m] = sm[0];
}

// Final heads: gd MLP with BN over batch axis, ld linear, fusion MLP.
__global__ void head_kernel(const float* ldmax, const float* gf,
                            const float* ld_w3, const float* ld_b3,
                            const float* gd_w1, const float* gd_b1,
                            const float* gd_w2, const float* gd_b2,
                            const float* fu_w1, const float* fu_b1,
                            const float* fu_w2, const float* fu_b2,
                            float* out) {
  __shared__ float g[2][128];
  __shared__ float lv[2], gv[2];
  int tid = threadIdx.x;
  if (tid < 256) {
    int b = tid >> 7, o = tid & 127;
    float s = gd_b1[o];
    for (int i = 0; i < 256; i++) s += gf[b * 256 + i] * gd_w1[o * 256 + i];
    g[b][o] = s;
  }
  __syncthreads();
  if (tid < 128) {
    float a = g[0][tid], c = g[1][tid];
    float m = 0.5f * (a + c);
    float v = 0.5f * ((a - m) * (a - m) + (c - m) * (c - m));
    float r = rsqrtf(v + BN_EPS);
    float na = (a - m) * r, nc = (c - m) * r;
    g[0][tid] = na > 0.f ? na : 0.2f * na;
    g[1][tid] = nc > 0.f ? nc : 0.2f * nc;
  }
  __syncthreads();
  if (tid < 2) {
    float s = gd_b2[0];
    for (int o = 0; o < 128; o++) s += g[tid][o] * gd_w2[o];
    gv[tid] = s;
    float l = ld_b3[0];
    for (int c = 0; c < 64; c++) l += ldmax[tid * 64 + c] * ld_w3[c];
    lv[tid] = l;
  }
  __syncthreads();
  if (tid < 2) {
    float c0 = lv[tid], c1 = gv[tid];
    float o = fu_b2[0];
    for (int h = 0; h < 16; h++) {
      float t = fu_b1[h] + c0 * fu_w1[h * 2 + 0] + c1 * fu_w1[h * 2 + 1];
      t = t > 0.f ? t : 0.2f * t;
      o += t * fu_w2[h];
    }
    out[tid] = o;
  }
}

// ---------------------------------------------------------------------------
// Host orchestration
// ---------------------------------------------------------------------------
extern "C" void kernel_launch(void* const* d_in, const int* in_sizes, int n_in,
                              void* d_out, int out_size, void* d_ws,
                              size_t ws_size, hipStream_t stream) {
  (void)in_sizes; (void)n_in; (void)out_size; (void)ws_size;
  const float* pts    = (const float*)d_in[0];   // (B, N, 3) point-major
  const float* ec1_w1 = (const float*)d_in[1];
  const float* ec1_w2 = (const float*)d_in[2];
  const float* ec1_w3 = (const float*)d_in[3];
  const float* ec2_w1 = (const float*)d_in[4];
  const float* ec2_w2 = (const float*)d_in[5];
  const float* ec2_w3 = (const float*)d_in[6];
  const float* ec3_w1 = (const float*)d_in[7];
  const float* ec3_w2 = (const float*)d_in[8];
  const float* ec3_w3 = (const float*)d_in[9];
  const float* sa1_w1 = (const float*)d_in[10];
  const float* sa1_w2 = (const float*)d_in[12];
  const float* sa1_b2 = (const float*)d_in[13];
  const float* sa2_w1 = (const float*)d_in[14];
  const float* sa2_w2 = (const float*)d_in[16];
  const float* sa2_b2 = (const float*)d_in[17];
  const float* sa3_w1 = (const float*)d_in[18];
  const float* sa3_w2 = (const float*)d_in[20];
  const float* sa3_b2 = (const float*)d_in[21];
  const float* ld_w1  = (const float*)d_in[22];
  const float* ld_w2  = (const float*)d_in[24];
  const float* ld_w3  = (const float*)d_in[26];
  const float* ld_b3  = (const float*)d_in[27];
  const float* gd_w1  = (const float*)d_in[28];
  const float* gd_b1  = (const float*)d_in[29];
  const float* gd_w2  = (const float*)d_in[30];
  const float* gd_b2  = (const float*)d_in[31];
  const float* fu_w1  = (const float*)d_in[32];
  const float* fu_b1  = (const float*)d_in[33];
  const float* fu_w2  = (const float*)d_in[34];
  const float* fu_b2  = (const float*)d_in[35];

  // ---- workspace carve-up (~100 MB), all point-major ----
  char* w = (char*)d_ws;
  size_t off = 0;
  auto take = [&](size_t bytes) -> void* {
    void* p = w + off;
    off = (off + bytes + 255) & ~(size_t)255;
    return p;
  };
  float*  FA   = (float*)take((size_t)Bc * Np * 256 * 4);  // feature ping
  float*  FB   = (float*)take((size_t)Bc * Np * 256 * 4);  // feature pong
  float*  NORM = (float*)take((size_t)Bc * Np * 4);
  int*    IDX  = (int*)take((size_t)Bc * Np * 16 * 4);
  half_t* H1   = (half_t*)take((size_t)64 * 262144 * 2);
  half_t* H2   = (half_t*)take((size_t)64 * 262144 * 2);
  float*  MV   = (float*)take(512 * 4);
  float*  MIND = (float*)take((size_t)Bc * Np * 4);
  int*    FIDX = (int*)take((size_t)Bc * 512 * 4);
  float*  NX1  = (float*)take((size_t)Bc * 512 * 3 * 4);
  float*  NX2  = (float*)take((size_t)Bc * 128 * 3 * 4);
  int*    SIDX = (int*)take((size_t)Bc * 512 * 32 * 4);
  float*  F1   = (float*)take((size_t)Bc * 512 * 128 * 4);
  float*  F2   = (float*)take((size_t)Bc * 128 * 256 * 4);
  float*  GF   = (float*)take((size_t)Bc * 256 * 4);
  float*  LDMX = (float*)take((size_t)Bc * 64 * 4);

  auto fillf = [&](float* p, size_t n, float v) {
    int blocks = (int)((n + 255) / 256);
    if (blocks > 4096) blocks = 4096;
    fill_kernel<<<blocks, 256, 0, stream>>>(p, n, v);
  };
  auto bn = [&](half_t* H, int M, int P, int act) {
    fillf(MV, (size_t)2 * M, 0.f);
    size_t total = (size_t)M * P;
    int blocks = (int)((total + 255) / 256);
    if (blocks > 1024) blocks = 1024;
    bn_stats_kernel<<<blocks, 256, 0, stream>>>(H, M, total, MV);
    int blocks2 = (int)((total + 255) / 256);
    if (blocks2 > 65535) blocks2 = 65535;
    bn_apply_kernel<<<blocks2, 256, 0, stream>>>(H, M, P, MV, act);
  };

  // ================= EdgeConv 1: 3 -> 64, k=16 (pts -> FB) ================
  knn3_kernel<<<(Bc * Np + 127) / 128, 128, 0, stream>>>(pts, Np, pts, Np, 16, IDX);
  {
    GemmArgs a{}; a.W = ec1_w1; a.M = 32; a.K = 6; a.P = Bc * Np * 16;
    a.srcA = pts; a.C = 3; a.N = Np; a.kE = 16; a.idx = IDX; a.outH = H1;
    gemm_wmma_kernel<1, 0><<<dim3(a.P / 64, a.M / 16), 128, 0, stream>>>(a);
    bn(H1, 32, a.P, 2);
    GemmArgs b{}; b.W = ec1_w2; b.M = 32; b.K = 32; b.P = a.P;
    b.srcH = H1; b.outH = H2;
    gemm_wmma_kernel<0, 0><<<dim3(b.P / 64, b.M / 16), 128, 0, stream>>>(b);
    bn(H2, 32, b.P, 2);
    fillf(FB, (size_t)Bc * Np * 64, -3.0e38f);
    GemmArgs c{}; c.W = ec1_w3; c.M = 64; c.K = 32; c.P = a.P;
    c.srcH = H2; c.outF = FB; c.Nout = Np; c.g = 16;
    gemm_wmma_kernel<0, 1><<<dim3(c.P / 64, c.M / 16), 128, 0, stream>>>(c);
  }

  // ================= EdgeConv 2: 64 -> 128, k=16 (FB -> FA) ===============
  norms_kernel<<<(Bc * Np + 255) / 256, 256, 0, stream>>>(FB, 64, Np, NORM);
  knn_wmma_kernel<<<dim3(Np / 16, Bc), 128, 0, stream>>>(FB, NORM, 64, Np, 16, IDX);
  {
    GemmArgs a{}; a.W = ec2_w1; a.M = 64; a.K = 128; a.P = Bc * Np * 16;
    a.srcA = FB; a.C = 64; a.N = Np; a.kE = 16; a.idx = IDX; a.outH = H1;
    gemm_wmma_kernel<1, 0><<<dim3(a.P / 64, a.M / 16), 128, 0, stream>>>(a);
    bn(H1, 64, a.P, 2);
    GemmArgs b{}; b.W = ec2_w2; b.M = 64; b.K = 64; b.P = a.P;
    b.srcH = H1; b.outH = H2;
    gemm_wmma_kernel<0, 0><<<dim3(b.P / 64, b.M / 16), 128, 0, stream>>>(b);
    bn(H2, 64, b.P, 2);
    fillf(FA, (size_t)Bc * Np * 128, -3.0e38f);
    GemmArgs c{}; c.W = ec2_w3; c.M = 128; c.K = 64; c.P = a.P;
    c.srcH = H2; c.outF = FA; c.Nout = Np; c.g = 16;
    gemm_wmma_kernel<0, 1><<<dim3(c.P / 64, c.M / 16), 128, 0, stream>>>(c);
  }

  // ================= EdgeConv 3: 128 -> 256, k=8 (FA -> FB = lf) ==========
  norms_kernel<<<(Bc * Np + 255) / 256, 256, 0, stream>>>(FA, 128, Np, NORM);
  knn_wmma_kernel<<<dim3(Np / 16, Bc), 128, 0, stream>>>(FA, NORM, 128, Np, 8, IDX);
  {
    GemmArgs a{}; a.W = ec3_w1; a.M = 128; a.K = 256; a.P = Bc * Np * 8;
    a.srcA = FA; a.C = 128; a.N = Np; a.kE = 8; a.idx = IDX; a.outH = H1;
    gemm_wmma_kernel<1, 0><<<dim3(a.P / 64, a.M / 16), 128, 0, stream>>>(a);
    bn(H1, 128, a.P, 2);
    GemmArgs b{}; b.W = ec3_w2; b.M = 128; b.K = 128; b.P = a.P;
    b.srcH = H1; b.outH = H2;
    gemm_wmma_kernel<0, 0><<<dim3(b.P / 64, b.M / 16), 128, 0, stream>>>(b);
    bn(H2, 128, b.P, 2);
    fillf(FB, (size_t)Bc * Np * 256, -3.0e38f);
    GemmArgs c{}; c.W = ec3_w3; c.M = 256; c.K = 128; c.P = a.P;
    c.srcH = H2; c.outF = FB; c.Nout = Np; c.g = 8;
    gemm_wmma_kernel<0, 1><<<dim3(c.P / 64, c.M / 16), 128, 0, stream>>>(c);
  }

  // ================= SA1: 8192 -> 512 pts, feats = xyz (C=3) ==============
  fps_kernel<<<Bc, 256, 0, stream>>>(pts, Np, 512, FIDX, MIND);
  gatherxyz_kernel<<<(Bc * 512 * 3 + 255) / 256, 256, 0, stream>>>(pts, Np, FIDX, 512, NX1);
  knn3_kernel<<<(Bc * 512 + 127) / 128, 128, 0, stream>>>(NX1, 512, pts, Np, 32, SIDX);
  {
    GemmArgs a{}; a.W = sa1_w1; a.M = 64; a.K = 6; a.P = Bc * 512 * 32;
    a.srcA = pts; a.srcB = pts; a.srcC = NX1; a.C = 3; a.N = Np; a.S = 512;
    a.kE = 32; a.idx = SIDX; a.outH = H1;
    gemm_wmma_kernel<2, 0><<<dim3(a.P / 64, a.M / 16), 128, 0, stream>>>(a);
    bn(H1, 64, a.P, 1);
    fillf(F1, (size_t)Bc * 512 * 128, -3.0e38f);
    GemmArgs b{}; b.W = sa1_w2; b.bias = sa1_b2; b.M = 128; b.K = 64;
    b.P = a.P; b.srcH = H1; b.outF = F1; b.Nout = 512; b.g = 32;
    gemm_wmma_kernel<0, 1><<<dim3(b.P / 64, b.M / 16), 128, 0, stream>>>(b);
  }

  // ================= SA2: 512 -> 128 pts, feats = F1 (C=128) ==============
  fps_kernel<<<Bc, 256, 0, stream>>>(NX1, 512, 128, FIDX, MIND);
  gatherxyz_kernel<<<(Bc * 128 * 3 + 255) / 256, 256, 0, stream>>>(NX1, 512, FIDX, 128, NX2);
  knn3_kernel<<<(Bc * 128 + 127) / 128, 128, 0, stream>>>(NX2, 128, NX1, 512, 32, SIDX);
  {
    GemmArgs a{}; a.W = sa2_w1; a.M = 256; a.K = 131; a.P = Bc * 128 * 32;
    a.srcA = NX1; a.srcB = F1; a.srcC = NX2; a.C = 128; a.N = 512; a.S = 128;
    a.kE = 32; a.idx = SIDX; a.outH = H1;
    gemm_wmma_kernel<2, 0><<<dim3(a.P / 64, a.M / 16), 128, 0, stream>>>(a);
    bn(H1, 256, a.P, 1);
    fillf(F2, (size_t)Bc * 128 * 256, -3.0e38f);
    GemmArgs b{}; b.W = sa2_w2; b.bias = sa2_b2; b.M = 256; b.K = 256;
    b.P = a.P; b.srcH = H1; b.outF = F2; b.Nout = 128; b.g = 32;
    gemm_wmma_kernel<0, 1><<<dim3(b.P / 64, b.M / 16), 128, 0, stream>>>(b);
  }

  // ================= SA3 (group-all): gf (B,256) ==========================
  {
    GemmArgs a{}; a.W = sa3_w1; a.M = 256; a.K = 259; a.P = Bc * 128;
    a.srcA = NX2; a.srcB = F2; a.C = 256; a.S = 128; a.outH = H1;
    gemm_wmma_kernel<3, 0><<<dim3(a.P / 64, a.M / 16), 128, 0, stream>>>(a);
    bn(H1, 256, a.P, 1);
    fillf(GF, (size_t)Bc * 256, -3.0e38f);
    GemmArgs b{}; b.W = sa3_w2; b.bias = sa3_b2; b.M = 256; b.K = 256;
    b.P = a.P; b.srcH = H1; b.outF = GF; b.Nout = 1; b.g = 128;
    gemm_wmma_kernel<0, 1><<<dim3(b.P / 64, b.M / 16), 128, 0, stream>>>(b);
  }

  // ================= Local discriminator head on lf = FB ==================
  {
    GemmArgs a{}; a.W = ld_w1; a.M = 128; a.K = 256; a.P = Bc * Np;
    a.srcA = FB; a.N = Np; a.outH = H1;
    gemm_wmma_kernel<4, 0><<<dim3(a.P / 64, a.M / 16), 128, 0, stream>>>(a);
    bn(H1, 128, a.P, 2);
    GemmArgs b{}; b.W = ld_w2; b.M = 64; b.K = 128; b.P = a.P;
    b.srcH = H1; b.outH = H2;
    gemm_wmma_kernel<0, 0><<<dim3(b.P / 64, b.M / 16), 128, 0, stream>>>(b);
    bn(H2, 64, b.P, 2);
    rowmax_kernel<<<dim3(64, Bc), 256, 0, stream>>>(H2, 64, Np, LDMX);
  }

  // ================= Global head + fusion =================================
  head_kernel<<<1, 256, 0, stream>>>(LDMX, GF, ld_w3, ld_b3, gd_w1, gd_b1,
                                     gd_w2, gd_b2, fu_w1, fu_b1, fu_w2, fu_b2,
                                     (float*)d_out);
}
